// AttentionBase_36421322670785
// MI455X (gfx1250) — matrix operand order repaired and checked by reading, not compiled, not run on hardware
//
#include <hip/hip_runtime.h>

#define NN  9216
#define CC  256
#define HHX 96
#define WWX 96

typedef float    v8f  __attribute__((ext_vector_type(8)));
typedef float    v2f  __attribute__((ext_vector_type(2)));
typedef _Float16 v16h __attribute__((ext_vector_type(16)));
typedef _Float16 v8h  __attribute__((ext_vector_type(8)));

// ---- helpers -------------------------------------------------------------

// 16 contiguous halves -> v16h (A-matrix per-lane payload; caller adds half*16)
static __device__ inline v16h load16h(const _Float16* p) {
  union { uint4 u[2]; v16h h; } u;
  u.u[0] = *(const uint4*)(p);
  u.u[1] = *(const uint4*)(p + 8);
  return u.h;
}
// 16-bit A-matrix 16x32: lane holds K {hb..hb+7} and {hb+16..hb+23}, hb=half*8
static __device__ inline v16h loadA16x32(const _Float16* row, int half) {
  const _Float16* p = row + half * 8;
  union { uint4 u[2]; v16h h; } u;
  u.u[0] = *(const uint4*)(p);
  u.u[1] = *(const uint4*)(p + 16);
  return u.h;
}
// lane <-> lane^16 exchange (group-of-32 swizzle, xor_mask=0x10, and_mask=0x1f)
static __device__ inline float swz16(float v) {
  int i = __builtin_bit_cast(int, v);
  i = __builtin_amdgcn_ds_swizzle(i, 0x401F);
  return __builtin_bit_cast(float, i);
}
static __device__ inline unsigned pack2h(float a, float b) {
  unsigned short ua = __builtin_bit_cast(unsigned short, (_Float16)a);
  unsigned short ub = __builtin_bit_cast(unsigned short, (_Float16)b);
  return (unsigned)ua | ((unsigned)ub << 16);
}

// ---- kernel 1: Q/K/V 1x1-conv projections via fp32 WMMA ------------------
// 20 output tiles (2 Q, 2 K, 16 V) x 576 n-tiles; one wave per 16x16 tile.
// Qh[N][32] f16 (WMMA-B layout source), Kpt[N][32] f16 = (K+pos)^T,
// Vh[C][N] f16 (m-contiguous: WMMA-A source for the PV GEMM).
__global__ __launch_bounds__(128) void proj_kernel(
    const float* __restrict__ x,  const float* __restrict__ xe,
    const float* __restrict__ qw, const float* __restrict__ qb,
    const float* __restrict__ kw, const float* __restrict__ kb,
    const float* __restrict__ vw, const float* __restrict__ vb,
    const float* __restrict__ hp, const float* __restrict__ wp,
    _Float16* __restrict__ Qh, _Float16* __restrict__ Kpt,
    _Float16* __restrict__ Vh)
{
  const int lane = threadIdx.x & 31;
  const int wid  = (blockIdx.x << 2) | (threadIdx.x >> 5);
  const int ot   = wid % 20;          // output tile (o / c dimension)
  const int nt   = wid / 20;          // pixel tile
  const int l15  = lane & 15;
  const int half = lane >> 4;
  const int n    = nt * 16 + l15;

  const float* W; const float* bias; const float* src; int o0;
  if (ot < 2)      { W = qw; bias = qb; src = x;  o0 = ot * 16; }
  else if (ot < 4) { W = kw; bias = kb; src = xe; o0 = (ot - 2) * 16; }
  else             { W = vw; bias = vb; src = xe; o0 = (ot - 4) * 16; }

  const float* wrow = W + (o0 + l15) * CC + half * 2;   // A: row o, K=k+half*2
  const float* bcol = src + (size_t)(half * 2) * NN + n;// B: col n, K=k+half*2

  v8f acc = {};
#pragma unroll 8
  for (int k = 0; k < CC; k += 4) {
    v2f a, b;
    a[0] = wrow[k];
    a[1] = wrow[k + 1];
    b[0] = bcol[(size_t)k * NN];
    b[1] = bcol[(size_t)(k + 1) * NN];
    acc = __builtin_amdgcn_wmma_f32_16x16x4_f32(false, a, false, b,
                                                (short)0, acc, false, false);
  }

  const int orow0 = o0 + half * 8;    // D rows: o = orow0 + j
  float g[8];
#pragma unroll
  for (int j = 0; j < 8; ++j) g[j] = acc[j] + bias[orow0 + j];

  if (ot < 4) {
    if (ot >= 2) {  // K path: add positional bias before f16 conversion
      const int hh = n / WWX, ww = n % WWX;
#pragma unroll
      for (int j = 0; j < 8; ++j) {
        const int c8 = orow0 + j;
        g[j] += hp[c8 * HHX + hh] + wp[c8 * WWX + ww];
      }
    }
    _Float16* dst = (ot < 2) ? Qh : Kpt;
    v8h hv;
#pragma unroll
    for (int j = 0; j < 8; ++j) hv[j] = (_Float16)g[j];
    *(v8h*)(dst + n * 32 + orow0) = hv;   // transposed store is contiguous
  } else {
#pragma unroll
    for (int j = 0; j < 8; ++j)
      Vh[(size_t)(orow0 + j) * NN + n] = (_Float16)g[j];
  }
}

// ---- kernel 2: fused flash attention + epilogue --------------------------
// Block = 4 waves, handles 16 query pixels; wave w owns channels [w*64,w*64+64).
// Per 64-key tile: S^T[m][q] via 4 WMMAs (A=Kpt, B=Q), online softmax with
// within-lane reductions + one xor-16 swizzle, O^T[c][q] += Vh(A) x P(B).
__global__ __launch_bounds__(128) void attn_kernel(
    const _Float16* __restrict__ Qh, const _Float16* __restrict__ Kpt,
    const _Float16* __restrict__ Vh, const float* __restrict__ x,
    const float* __restrict__ gamma, float* __restrict__ out)
{
  const int lane  = threadIdx.x & 31;
  const int wave  = threadIdx.x >> 5;
  const int l15   = lane & 15;
  const int half  = lane >> 4;
  const int q0    = blockIdx.x * 16;
  const int cbase = wave * 64;

  // Q as B-matrix (K=c8 contiguous in Qh[N][32]); loaded once, kept in regs
  const v16h Qb = loadA16x32(Qh + (q0 + l15) * 32, half);

  v8f acc[4] = {};
  float mrun = -1.0e30f;
  float lrun = 0.0f;

  for (int m0 = 0; m0 < NN; m0 += 64) {
    const v8f z = {};
    v8f s[4];
#pragma unroll
    for (int t = 0; t < 4; ++t) {
      v16h a = loadA16x32(Kpt + (m0 + t * 16 + l15) * 32, half);
      s[t] = __builtin_amdgcn_wmma_f32_16x16x32_f16(false, a, false, Qb,
                                                    (short)0, z, false, false);
    }

    // online softmax over the key axis (all 32 per-lane values share q=l15)
    float tmax = s[0][0];
#pragma unroll
    for (int t = 0; t < 4; ++t)
#pragma unroll
      for (int j = 0; j < 8; ++j) tmax = fmaxf(tmax, s[t][j]);
    tmax = fmaxf(tmax, swz16(tmax));
    const float mnew  = fmaxf(mrun, tmax);
    const float alpha = __expf(mrun - mnew);
    mrun = mnew;

    float rsum = 0.0f;
#pragma unroll
    for (int t = 0; t < 4; ++t)
#pragma unroll
      for (int j = 0; j < 8; ++j) {
        const float p = __expf(s[t][j] - mnew);
        s[t][j] = p;
        rsum += p;
      }
    rsum += swz16(rsum);
    lrun = lrun * alpha + rsum;

#pragma unroll
    for (int ct = 0; ct < 4; ++ct)
#pragma unroll
      for (int j = 0; j < 8; ++j) acc[ct][j] *= alpha;

    // P tiles (S^T layout) -> B-matrix chunks of K=32 keys; xor-16 exchange
#pragma unroll
    for (int ch = 0; ch < 2; ++ch) {
      const int t0 = ch * 2;
      float seq[16];
#pragma unroll
      for (int j = 0; j < 8; ++j) {
        const float sw0 = swz16(s[t0][j]);
        const float sw1 = swz16(s[t0 + 1][j]);
        seq[j]     = half ? sw1            : s[t0][j];   // K = local m 0..7 / 16..23
        seq[8 + j] = half ? s[t0 + 1][j]   : sw0;        // K = local m 8..15 / 24..31
      }
      union { unsigned u[8]; v16h h; } pb;
#pragma unroll
      for (int p = 0; p < 8; ++p) pb.u[p] = pack2h(seq[2 * p], seq[2 * p + 1]);

#pragma unroll
      for (int ct = 0; ct < 4; ++ct) {
        v16h av = load16h(
            Vh + (size_t)(cbase + ct * 16 + l15) * NN + m0 + ch * 32 + half * 16);
        acc[ct] = __builtin_amdgcn_wmma_f32_16x16x32_f16(
            false, av, false, pb.h, (short)0, acc[ct], false, false);
      }
    }
  }

  // epilogue: out = gamma * (acc / l) + x
  const float inv = 1.0f / lrun;
  const float gm  = gamma[0];
  const int n = q0 + l15;
#pragma unroll
  for (int ct = 0; ct < 4; ++ct)
#pragma unroll
    for (int j = 0; j < 8; ++j) {
      const size_t idx = (size_t)(cbase + ct * 16 + half * 8 + j) * NN + n;
      out[idx] = gm * (acc[ct][j] * inv) + x[idx];
    }
}

// ---- launcher ------------------------------------------------------------
extern "C" void kernel_launch(void* const* d_in, const int* in_sizes, int n_in,
                              void* d_out, int out_size, void* d_ws, size_t ws_size,
                              hipStream_t stream) {
  const float* x  = (const float*)d_in[0];
  const float* xe = (const float*)d_in[1];
  const float* qw = (const float*)d_in[2];
  const float* qb = (const float*)d_in[3];
  const float* kw = (const float*)d_in[4];
  const float* kb = (const float*)d_in[5];
  const float* vw = (const float*)d_in[6];
  const float* vb = (const float*)d_in[7];
  const float* hp = (const float*)d_in[8];
  const float* wp = (const float*)d_in[9];
  const float* gm = (const float*)d_in[10];

  _Float16* Qh  = (_Float16*)d_ws;                 // [N][32] f16
  _Float16* Kpt = Qh  + (size_t)NN * 32;           // [N][32] f16
  _Float16* Vh  = Kpt + (size_t)NN * 32;           // [C][N] f16  (~5.9 MB total)

  // 20 o-tiles * 576 n-tiles = 11520 wave-tiles / 4 waves per block
  proj_kernel<<<2880, 128, 0, stream>>>(x, xe, qw, qb, kw, kb, vw, vb, hp, wp,
                                        Qh, Kpt, Vh);
  // 576 query tiles of 16 pixels, 4 waves each (channel split)
  attn_kernel<<<576, 128, 0, stream>>>(Qh, Kpt, Vh, x, gm, (float*)d_out);
}
